// GatedSelfAttention_70471823392817
// MI455X (gfx1250) — matrix operand order repaired and checked by reading, not compiled, not run on hardware
//
#include <hip/hip_runtime.h>

// ---------------------------------------------------------------------------
// Types / helpers
// ---------------------------------------------------------------------------
typedef __attribute__((ext_vector_type(8)))  float  v8f;
typedef __attribute__((ext_vector_type(16))) __bf16 v16bf;

union Frag32B { v16bf b; uint4 q[2]; };
union Pack8   { unsigned short s[8]; uint4 q; };

static __device__ __forceinline__ unsigned short f2bfu(float f) {
    unsigned u = __builtin_bit_cast(unsigned, f);
    unsigned r = u + 0x7FFFu + ((u >> 16) & 1u);   // round-to-nearest-even
    return (unsigned short)(r >> 16);
}

// fragment = two 16-byte contiguous blocks
static __device__ __forceinline__ v16bf load2x16(const unsigned short* p0,
                                                 const unsigned short* p1) {
    Frag32B f;
    f.q[0] = *(const uint4*)p0;
    f.q[1] = *(const uint4*)p1;
    return f.b;
}
// fragment = one 32-byte contiguous block
static __device__ __forceinline__ v16bf load32(const unsigned short* p) {
    Frag32B f;
    f.q[0] = ((const uint4*)p)[0];
    f.q[1] = ((const uint4*)p)[1];
    return f.b;
}

// ---- DPP 16-lane reductions (pure VALU, no LDS): xor1, xor2, half-mirror,
// ---- mirror fully reduce each 16-lane group (C-matrix row-half span).
template <int CTRL>
static __device__ __forceinline__ float dpp_perm(float x) {
    int v = __builtin_amdgcn_mov_dpp(__builtin_bit_cast(int, x), CTRL, 0xF, 0xF, true);
    return __builtin_bit_cast(float, v);
}
static __device__ __forceinline__ float redmax16(float x) {
    x = fmaxf(x, dpp_perm<0xB1>(x));    // quad_perm [1,0,3,2]  : xor 1
    x = fmaxf(x, dpp_perm<0x4E>(x));    // quad_perm [2,3,0,1]  : xor 2
    x = fmaxf(x, dpp_perm<0x141>(x));   // row_half_mirror      : combine quads
    x = fmaxf(x, dpp_perm<0x140>(x));   // row_mirror           : combine halves
    return x;
}
static __device__ __forceinline__ float redsum16(float x) {
    x += dpp_perm<0xB1>(x);
    x += dpp_perm<0x4E>(x);
    x += dpp_perm<0x141>(x);
    x += dpp_perm<0x140>(x);
    return x;
}

static constexpr int B_  = 16;
static constexpr int LW_ = 512;
static constexpr int LE_ = 128;
static constexpr int L_  = 640;
static constexpr int H_  = 16;
static constexpr int DH_ = 64;
static constexpr int D_  = 1024;

// ---------------------------------------------------------------------------
// Kernel 1: projection GEMM  (X @ W + bias) -> bf16 workspace
//   grid: (tilesM_per_batch, 8, B).  Tile 64(M) x 128(N), 8 waves, 32x32/wave.
//   OUTT==0: out[b,h,l,dh]   (rows of Q/K/gate tensors)
//   OUTT==1: out[b,h,dh,l]   (V stored transposed for P*V B-fragments)
// ---------------------------------------------------------------------------
template <int OUTT, int LEN0, int LEN1>
__global__ __launch_bounds__(256)
void proj_gemm_bf16(const float* __restrict__ src0, const float* __restrict__ src1,
                    const float* __restrict__ W, const float* __restrict__ bias,
                    unsigned short* __restrict__ out) {
    constexpr int TK   = 32;
    constexpr int LOUT = LEN0 + LEN1;
    __shared__ unsigned short As[64][TK];     // [m][k]      4 KB
    __shared__ unsigned short BsT[128][TK];   // [n][k]      8 KB (pre-transposed)

    const int tid   = threadIdx.x;
    const int lane  = tid & 31;
    const int wave  = tid >> 5;
    const int wm    = wave >> 2;                  // 0..1
    const int wn    = wave & 3;                   // 0..3
    const unsigned bi    = blockIdx.z;
    const unsigned rowB  = blockIdx.x * 64;       // row offset within batch
    const unsigned tileN = blockIdx.y * 128;
    const int lh    = (lane >= 16) ? 1 : 0;
    const int l15   = lane & 15;
    const int kb8   = lh * 8;
    const int kb16  = lh * 16;

    // A staging map: row = tid>>2 (0..63), 8 consecutive floats
    const int arow = tid >> 2;
    const int acol = (tid & 3) * 8;
    const unsigned li = rowB + arow;
    const float* arow_ptr;
    if constexpr (LEN1 > 0) {
        arow_ptr = (li < (unsigned)LEN0)
            ? (src0 + (size_t)(bi * LEN0 + li) * D_)
            : (src1 + (size_t)(bi * LEN1 + (li - LEN0)) * D_);
    } else {
        arow_ptr = src0 + (size_t)(bi * LEN0 + li) * D_;
    }

    // B staging map: k-row = tid>>3 (0..31), 16 consecutive floats
    const int brow = tid >> 3;
    const int bcol = (tid & 7) * 16;
    const float* wrow_ptr = W + (size_t)brow * D_ + tileN + bcol;

    v8f acc[2][2];
    #pragma unroll
    for (int a = 0; a < 2; ++a)
        #pragma unroll
        for (int c = 0; c < 2; ++c)
            acc[a][c] = v8f{0.f,0.f,0.f,0.f,0.f,0.f,0.f,0.f};

    for (int k0 = 0; k0 < D_; k0 += TK) {
        // ---- stage A: 2 x global b128, pack bf16, 1 x lds b128 store
        {
            const float4* a4 = (const float4*)(arow_ptr + k0 + acol);
            float4 t0 = a4[0], t1 = a4[1];
            Pack8 p;
            p.s[0] = f2bfu(t0.x); p.s[1] = f2bfu(t0.y);
            p.s[2] = f2bfu(t0.z); p.s[3] = f2bfu(t0.w);
            p.s[4] = f2bfu(t1.x); p.s[5] = f2bfu(t1.y);
            p.s[6] = f2bfu(t1.z); p.s[7] = f2bfu(t1.w);
            *(uint4*)&As[arow][acol] = p.q;
        }
        // ---- stage B transposed: 4 x global b128, 16 scattered b16 stores
        {
            const float4* w4 = (const float4*)(wrow_ptr + (size_t)k0 * D_);
            #pragma unroll
            for (int j = 0; j < 4; ++j) {
                float4 t = w4[j];
                BsT[bcol + j*4 + 0][brow] = f2bfu(t.x);
                BsT[bcol + j*4 + 1][brow] = f2bfu(t.y);
                BsT[bcol + j*4 + 2][brow] = f2bfu(t.z);
                BsT[bcol + j*4 + 3][brow] = f2bfu(t.w);
            }
        }
        __syncthreads();

        // ---- fragments: all vector LDS loads (contiguous per lane)
        v16bf af[2], bfr[2];
        #pragma unroll
        for (int ms = 0; ms < 2; ++ms) {
            const int row = wm * 32 + ms * 16 + l15;
            af[ms] = load32(&As[row][kb16]);
        }
        #pragma unroll
        for (int ns = 0; ns < 2; ++ns) {
            const int col = wn * 32 + ns * 16 + l15;
            bfr[ns] = load32(&BsT[col][kb16]);
        }
        #pragma unroll
        for (int ms = 0; ms < 2; ++ms)
            #pragma unroll
            for (int ns = 0; ns < 2; ++ns)
                acc[ms][ns] = __builtin_amdgcn_wmma_f32_16x16x32_bf16(
                    false, af[ms], false, bfr[ns], (short)0, acc[ms][ns], false, false);
        __syncthreads();
    }

    // ---- epilogue: +bias, scatter bf16 (all-constant strides, 32-bit math)
    #pragma unroll
    for (int ms = 0; ms < 2; ++ms) {
        #pragma unroll
        for (int ns = 0; ns < 2; ++ns) {
            const unsigned cg = tileN + wn * 32 + ns * 16 + l15;
            const unsigned hd = cg >> 6, dh = cg & 63;
            const float bv = bias[cg];
            const unsigned bh = bi * H_ + hd;
            #pragma unroll
            for (int r = 0; r < 8; ++r) {
                const unsigned ol = rowB + wm * 32 + ms * 16 + r + 8 * lh;
                const float vv = acc[ms][ns][r] + bv;
                const unsigned idx = OUTT
                    ? (bh * DH_ + dh) * LOUT + ol      // [B,H,DH,L]
                    : (bh * LOUT + ol) * DH_ + dh;     // [B,H,L,DH]
                out[idx] = f2bfu(vv);
            }
        }
    }
}

// ---------------------------------------------------------------------------
// Kernel 2: flash-style gated attention.
//   ONE WAVE PER BLOCK (32 threads): qt = blockIdx.x is an SGPR, so the
//   word/entity split and the per-chunk cross-block test are scalar branches
//   (no EXEC save/restore, no vector selects).
//   Operand layouts make every fragment a contiguous vector load:
//     Q / gateQ : [*,64] rows          -> A-frag = 2 x b128 global
//     K / gateK : [key][64]            -> B-frag = 1 x 32B contiguous global
//     V         : transposed [dh][L]   -> B-frag = 1 x 32B contiguous global
//     P         : LDS bounce [16][32]  -> A-frag = 2 x b128 LDS
//   Softmax row reductions use DPP (no LDS traffic).
// ---------------------------------------------------------------------------
__global__ __launch_bounds__(32)
void gated_attn(const unsigned short* __restrict__ q,  const unsigned short* __restrict__ k,
                const unsigned short* __restrict__ vt,
                const unsigned short* __restrict__ w2e_q, const unsigned short* __restrict__ w2e_k,
                const unsigned short* __restrict__ e2w_q, const unsigned short* __restrict__ e2w_k,
                const float* __restrict__ mask, float* __restrict__ out) {
    __shared__ unsigned short ldsP[16][32];      // P bounce, 1 KB

    const int lane = threadIdx.x;                // 0..31
    const int qt   = blockIdx.x;                 // 0..39  (scalar!)
    const unsigned h = blockIdx.y;
    const unsigned b = blockIdx.z;
    const int qrow0 = qt * 16;
    const bool q_word = (qrow0 < LW_);           // scalar condition
    const int lh   = (lane >= 16) ? 1 : 0;
    const int l15  = lane & 15;
    const int kb8  = lh * 8;
    const int kb16 = lh * 16;

    const unsigned bh = b * H_ + h;
    const unsigned short* qp  = q  + (unsigned)(bh * L_ + qrow0) * DH_;
    const unsigned short* kp  = k  + (unsigned)bh * (L_ * DH_);
    const unsigned short* vtp = vt + (unsigned)bh * (DH_ * L_);
    const unsigned short* gqp = q_word ? (w2e_q + (unsigned)(bh * LW_ + qrow0) * DH_)
                                       : (e2w_q + (unsigned)(bh * LE_ + (qrow0 - LW_)) * DH_);
    const unsigned short* gkw = w2e_k + (unsigned)bh * (LE_ * DH_); // keys = entities (rel)
    const unsigned short* gke = e2w_k + (unsigned)bh * (LW_ * DH_); // keys = words

    // output base: whole tile is uniformly word or entity (512 % 16 == 0)
    float* op = q_word
        ? (out + (size_t)(b * LW_ + qrow0) * D_)
        : (out + (size_t)B_ * LW_ * D_ + (size_t)(b * LE_ + (qrow0 - LW_)) * D_);

    // Q and gate-Q A-fragments: direct global vector loads
    v16bf qa[2], ga[2];
    #pragma unroll
    for (int j = 0; j < 2; ++j) {
        const unsigned short* qr = qp  + (unsigned)l15 * DH_ + j * 32;
        const unsigned short* gr = gqp + (unsigned)l15 * DH_ + j * 32;
        qa[j] = load32(qr + kb16);
        ga[j] = load32(gr + kb16);
    }

    float m_i[8], l_i[8];
    #pragma unroll
    for (int r = 0; r < 8; ++r) { m_i[r] = -1e30f; l_i[r] = 0.f; }
    v8f O[4];
    #pragma unroll
    for (int f = 0; f < 4; ++f) O[f] = v8f{0.f,0.f,0.f,0.f,0.f,0.f,0.f,0.f};

    for (int kc = 0; kc < 20; ++kc) {
        const int krow0 = kc * 32;
        const bool k_word = (krow0 < LW_);       // scalar

        // ---- S = Q K^T for 32 keys (2 C frags, K=64 in two steps)
        v8f s0 = v8f{0.f,0.f,0.f,0.f,0.f,0.f,0.f,0.f};
        v8f s1 = s0;
        #pragma unroll
        for (int j = 0; j < 2; ++j) {
            s0 = __builtin_amdgcn_wmma_f32_16x16x32_bf16(false, qa[j], false,
                     load32(kp + (unsigned)(krow0      + l15) * DH_ + j * 32 + kb16),
                     (short)0, s0, false, false);
            s1 = __builtin_amdgcn_wmma_f32_16x16x32_bf16(false, qa[j], false,
                     load32(kp + (unsigned)(krow0 + 16 + l15) * DH_ + j * 32 + kb16),
                     (short)0, s1, false, false);
        }

        const float mv0 = mask[b * L_ + krow0      + l15];
        const float mv1 = mask[b * L_ + krow0 + 16 + l15];

        // ---- fold gate into the softmax input:
        //   same-region :  sv = s*0.125 + (mv - 0.625)               [gate = -5]
        //   cross-region:  sv = s*0.125 + (mv - 1.25*sigmoid(g))     [gate = -10*sig]
        float sv[2][8];
        if (k_word == q_word) {                  // scalar branch
            const float a0 = mv0 - 0.625f, a1 = mv1 - 0.625f;
            #pragma unroll
            for (int r = 0; r < 8; ++r) {
                sv[0][r] = fmaf(s0[r], 0.125f, a0);
                sv[1][r] = fmaf(s1[r], 0.125f, a1);
            }
        } else {
            const unsigned short* gkp = q_word ? gkw : gke;
            const int krel0 = q_word ? (krow0 - LW_) : krow0;
            v8f g0 = v8f{0.f,0.f,0.f,0.f,0.f,0.f,0.f,0.f};
            v8f g1 = g0;
            #pragma unroll
            for (int j = 0; j < 2; ++j) {
                g0 = __builtin_amdgcn_wmma_f32_16x16x32_bf16(false, ga[j], false,
                         load32(gkp + (unsigned)(krel0      + l15) * DH_ + j * 32 + kb16),
                         (short)0, g0, false, false);
                g1 = __builtin_amdgcn_wmma_f32_16x16x32_bf16(false, ga[j], false,
                         load32(gkp + (unsigned)(krel0 + 16 + l15) * DH_ + j * 32 + kb16),
                         (short)0, g1, false, false);
            }
            #pragma unroll
            for (int r = 0; r < 8; ++r) {
                const float sg0 = 1.25f / (1.f + __expf(-g0[r]));
                const float sg1 = 1.25f / (1.f + __expf(-g1[r]));
                sv[0][r] = fmaf(s0[r], 0.125f, mv0 - sg0);
                sv[1][r] = fmaf(s1[r], 0.125f, mv1 - sg1);
            }
        }

        // ---- online softmax: DPP reductions within 16-lane row-halves
        #pragma unroll
        for (int r = 0; r < 8; ++r) {
            const float mx   = redmax16(fmaxf(sv[0][r], sv[1][r]));
            const float mnew = fmaxf(m_i[r], mx);
            const float sc   = __expf(m_i[r] - mnew);
            const float p0   = __expf(sv[0][r] - mnew);
            const float p1   = __expf(sv[1][r] - mnew);
            const float rs   = redsum16(p0 + p1);
            l_i[r] = l_i[r] * sc + rs;
            m_i[r] = mnew;
            #pragma unroll
            for (int f = 0; f < 4; ++f) O[f][r] *= sc;
            ldsP[r + 8 * lh][l15]      = f2bfu(p0);
            ldsP[r + 8 * lh][16 + l15] = f2bfu(p1);
        }

        // ---- P (C-layout) -> A-fragment: 2 x b128 LDS loads
        const v16bf pa = load32(&ldsP[l15][kb16]);

        // ---- O += P * V  (V transposed: contiguous B-frags)
        #pragma unroll
        for (int f = 0; f < 4; ++f)
            O[f] = __builtin_amdgcn_wmma_f32_16x16x32_bf16(false, pa, false,
                       load32(vtp + (unsigned)(f * 16 + l15) * L_ + krow0 + kb16),
                       (short)0, O[f], false, false);
    }

    // ---- epilogue: one reciprocal per row, strided f32 stores
    #pragma unroll
    for (int r = 0; r < 8; ++r) {
        const float inv = 1.0f / l_i[r];
        const unsigned rowoff = (unsigned)(r + 8 * lh) * D_ + h * DH_ + l15;
        #pragma unroll
        for (int f = 0; f < 4; ++f)
            op[rowoff + f * 16] = O[f][r] * inv;
    }
}

// ---------------------------------------------------------------------------
// Host launcher
// ---------------------------------------------------------------------------
extern "C" void kernel_launch(void* const* d_in, const int* in_sizes, int n_in,
                              void* d_out, int out_size, void* d_ws, size_t ws_size,
                              hipStream_t stream) {
    const float* word   = (const float*)d_in[0];
    const float* entity = (const float*)d_in[1];
    const float* mask   = (const float*)d_in[2];
    const float* qw = (const float*)d_in[4];  const float* qb = (const float*)d_in[5];
    const float* kw = (const float*)d_in[6];  const float* kb = (const float*)d_in[7];
    const float* vw = (const float*)d_in[8];  const float* vb = (const float*)d_in[9];
    const float* w2e_qw = (const float*)d_in[10]; const float* w2e_qb = (const float*)d_in[11];
    const float* w2e_kw = (const float*)d_in[12]; const float* w2e_kb = (const float*)d_in[13];
    const float* e2w_qw = (const float*)d_in[14]; const float* e2w_qb = (const float*)d_in[15];
    const float* e2w_kw = (const float*)d_in[16]; const float* e2w_kb = (const float*)d_in[17];

    unsigned short* ws = (unsigned short*)d_ws;
    const size_t n_qkv = (size_t)B_ * H_ * L_  * DH_;   // 10,485,760
    const size_t n_w   = (size_t)B_ * H_ * LW_ * DH_;   //  8,388,608
    const size_t n_e   = (size_t)B_ * H_ * LE_ * DH_;   //  2,097,152
    unsigned short* qws  = ws;
    unsigned short* kws  = qws + n_qkv;
    unsigned short* vws  = kws + n_qkv;                 // stored transposed [B,H,DH,L]
    unsigned short* w2eq = vws + n_qkv;
    unsigned short* w2ek = w2eq + n_w;
    unsigned short* e2wq = w2ek + n_e;
    unsigned short* e2wk = e2wq + n_e;

    const dim3 blk(256);
    const dim3 gFull(L_  / 64, D_ / 128, B_);   // 10 x 8 x 16
    const dim3 gWord(LW_ / 64, D_ / 128, B_);   //  8 x 8 x 16
    const dim3 gEnt (LE_ / 64, D_ / 128, B_);   //  2 x 8 x 16

    proj_gemm_bf16<0, LW_, LE_><<<gFull, blk, 0, stream>>>(word, entity, qw, qb, qws);
    proj_gemm_bf16<0, LW_, LE_><<<gFull, blk, 0, stream>>>(word, entity, kw, kb, kws);
    proj_gemm_bf16<1, LW_, LE_><<<gFull, blk, 0, stream>>>(word, entity, vw, vb, vws);
    proj_gemm_bf16<0, LW_, 0><<<gWord, blk, 0, stream>>>(word, nullptr, w2e_qw, w2e_qb, w2eq);
    proj_gemm_bf16<0, LE_, 0><<<gEnt,  blk, 0, stream>>>(entity, nullptr, w2e_kw, w2e_kb, w2ek);
    proj_gemm_bf16<0, LE_, 0><<<gEnt,  blk, 0, stream>>>(entity, nullptr, e2w_qw, e2w_qb, e2wq);
    proj_gemm_bf16<0, LW_, 0><<<gWord, blk, 0, stream>>>(word, nullptr, e2w_kw, e2w_kb, e2wk);

    gated_attn<<<dim3(40, H_, B_), 32, 0, stream>>>(
        qws, kws, vws, w2eq, w2ek, e2wq, e2wk, mask, (float*)d_out);
}
